// Transformer_48644799594669
// MI455X (gfx1250) — compile-verified, matrix operand order checked by reading, not run on hardware
//
#include <hip/hip_runtime.h>
#include <math.h>

// ---------------- problem constants (from reference) ----------------
#define Hn     1024
#define NHn    16
#define HDn    64
#define SEQn   512
#define Bn     4
#define Mn     2048          // Bn*SEQn tokens
#define FFNn   8192          // 8*H
#define FFHALF 4096          // 4*H
#define VOUTn  32000
#define SCALEF 0.125f        // 1/sqrt(64)

typedef __attribute__((ext_vector_type(16))) __bf16 v16bf;
typedef __attribute__((ext_vector_type(8)))  __bf16 v8bf;
typedef __attribute__((ext_vector_type(8)))  float  v8f;
typedef __attribute__((ext_vector_type(4))) unsigned int u32x4;
typedef __attribute__((ext_vector_type(8)))  int i32x8;
typedef __attribute__((ext_vector_type(4)))  int i32x4;

#ifndef __has_builtin
#define __has_builtin(x) 0
#endif
#if __has_builtin(__builtin_amdgcn_tensor_load_to_lds) && \
    __has_builtin(__builtin_amdgcn_s_wait_tensorcnt)
#define USE_TDM 1
#else
#define USE_TDM 0
#endif

// ---- WMMA bf16 fragment loaders --------------------------------------------
// 16-bit A/B operand layout (CDNA5 ISA 7.12.2): lanes 0-15 = rows 0..15,
// lanes 16-31 same rows; lo-half lanes hold K=0..7 & 16..23, hi-half lanes hold
// K=8..15 & 24..31.  Caller pre-offsets pointer by (hi?8:0) elements.
__device__ __forceinline__ v16bf load_frag(const __bf16* __restrict__ p) {
  v8bf lo = *(const v8bf*)p;          // 16B aligned
  v8bf hv = *(const v8bf*)(p + 16);   // 16B aligned
  v16bf f;
#pragma unroll
  for (int i = 0; i < 8; ++i) { f[i] = lo[i]; f[i + 8] = hv[i]; }
  return f;
}

__device__ __forceinline__ v16bf load_frag_strided(const __bf16* __restrict__ p, int ld) {
  v16bf f;
#pragma unroll
  for (int i = 0; i < 8; ++i) {
    f[i]     = p[(size_t)i * ld];
    f[i + 8] = p[(size_t)(16 + i) * ld];
  }
  return f;
}

// ---------------- fp32 -> bf16 bulk conversion ------------------------------
__global__ void cvt_f32_bf16(const float* __restrict__ in, __bf16* __restrict__ out) {
  const size_t i = ((size_t)blockIdx.x * blockDim.x + threadIdx.x) * 4;
#pragma unroll
  for (int j = 0; j < 4; ++j) out[i + j] = (__bf16)in[i + j];
}

#if USE_TDM
// Issue a TDM load of a 64-row x 32-elem bf16 tile (row stride K elems) from
// global into LDS at lds_byte_off.  Hardware pads +16B per 64B row so the
// staged column stride is 80B (bank-conflict-free, 16B aligned).
#define LDS_COL_STRIDE 40   // bf16 elems per staged column (32 data + 8 pad)
__device__ __forceinline__ void tdm_load_tile(const __bf16* gsrc, unsigned lds_byte_off,
                                              int K, int Nrows) {
  const unsigned long long ga = (unsigned long long)(uintptr_t)gsrc;
  u32x4 g0;
  g0[0] = 1u;                                          // count=1 (valid user D#)
  g0[1] = lds_byte_off;                                // lds_addr (bytes)
  g0[2] = (unsigned)(ga & 0xFFFFFFFFull);              // global_addr[31:0]
  g0[3] = ((unsigned)(ga >> 32) & 0x01FFFFFFu)         // global_addr[56:32]
        | (2u << 30);                                  // type=2 ("image")
  i32x8 g1;
  g1[0] = (1 << 16)                                    // data_size = 2 bytes
        | (1 << 20)                                    // pad_enable
        | (3 << 22)                                    // pad_interval = 16 DWORDs
        | (3 << 25);                                   // pad_amount  = 4 DWORDs
  g1[1] = (K & 0xFFFF) << 16;                          // tensor_dim0[15:0]
  g1[2] = (int)(((unsigned)K >> 16) | (((unsigned)Nrows & 0xFFFFu) << 16));
  g1[3] = (int)(((unsigned)Nrows >> 16) | (32u << 16)); // tile_dim0 = 32
  g1[4] = 64;                                          // tile_dim1 = 64, tile_dim2 = 0
  g1[5] = K;                                           // tensor_dim0_stride[31:0]
  g1[6] = 0;
  g1[7] = 0;
  i32x4 z4 = {0, 0, 0, 0};
#if __clang_major__ >= 23
  i32x8 z8 = {0, 0, 0, 0, 0, 0, 0, 0};
  __builtin_amdgcn_tensor_load_to_lds(g0, g1, z4, z4, z8, 0);
#else
  __builtin_amdgcn_tensor_load_to_lds(g0, g1, z4, z4, 0);
#endif
}
#endif

// ---------------- generic GEMM: C[M,N] = A[M,K] @ W[N,K]^T + bias (+res) ----
// block = 256 threads (8 waves); block tile 128x64; wave tile 16x64 (4 wmma acc)
// W tile (64x32 bf16) staged via TDM into double-buffered LDS when available.
template <typename OutT>
__global__ void gemm_wmma(const __bf16* __restrict__ A, const __bf16* __restrict__ W,
                          const float* __restrict__ bias, const float* __restrict__ residual,
                          OutT* __restrict__ C, int M, int N, int K) {
  const int lane = threadIdx.x & 31;
  const int wave = threadIdx.x >> 5;
  const int hi = lane >> 4, lr = lane & 15;
  const int row = blockIdx.y * 128 + wave * 16 + lr;
  const int colBase = blockIdx.x * 64;

  v8f acc[4] = {};
#if USE_TDM
  __shared__ __bf16 bstage[2][64 * LDS_COL_STRIDE];
  const __bf16* wtile = W + (size_t)colBase * K;
  const int nk = K / 32;
  if (threadIdx.x < 32)
    tdm_load_tile(wtile, (unsigned)(uintptr_t)&bstage[0][0], K, N);
  for (int i = 0; i < nk; ++i) {
    const int cur = i & 1;
    if (threadIdx.x < 32)
      __builtin_amdgcn_s_wait_tensorcnt(0);   // buffer 'cur' filled
    __syncthreads();                          // publish to all waves
    if (threadIdx.x < 32 && i + 1 < nk)       // prefetch next tile (overlaps WMMA)
      tdm_load_tile(wtile + (i + 1) * 32, (unsigned)(uintptr_t)&bstage[cur ^ 1][0], K, N);
    v16bf af = load_frag(A + (size_t)row * K + i * 32 + hi * 8);
    const __bf16* bs = &bstage[cur][hi * 8];
#pragma unroll
    for (int t = 0; t < 4; ++t) {
      v16bf bf = load_frag(bs + (t * 16 + lr) * LDS_COL_STRIDE);
      acc[t] = __builtin_amdgcn_wmma_f32_16x16x32_bf16(false, af, false, bf,
                                                       (short)0, acc[t], false, false);
    }
    __syncthreads();                          // done reading 'cur' before its reuse
  }
#else
  for (int k0 = 0; k0 < K; k0 += 32) {
    v16bf af = load_frag(A + (size_t)row * K + k0 + hi * 8);
#pragma unroll
    for (int t = 0; t < 4; ++t) {
      v16bf bf = load_frag(W + (size_t)(colBase + t * 16 + lr) * K + k0 + hi * 8);
      acc[t] = __builtin_amdgcn_wmma_f32_16x16x32_bf16(false, af, false, bf,
                                                       (short)0, acc[t], false, false);
    }
  }
#endif
  const int crow = blockIdx.y * 128 + wave * 16 + hi * 8;  // D: lane-hi selects M+8
#pragma unroll
  for (int t = 0; t < 4; ++t) {
    const int col = colBase + t * 16 + lr;
    const float bv = bias ? bias[col] : 0.0f;
#pragma unroll
    for (int r = 0; r < 8; ++r) {
      const size_t idx = (size_t)(crow + r) * N + col;
      float v = acc[t][r] + bv;
      if (residual) v += residual[idx];
      C[idx] = (OutT)v;
    }
  }
}

// ---------------- attention scores: S[bh,q,k] = scale * Q.Kt, masked --------
__global__ void scores_wmma(const __bf16* __restrict__ Q, const __bf16* __restrict__ Km,
                            float* __restrict__ S, const unsigned char* __restrict__ kpad,
                            int causal) {
  const int lane = threadIdx.x & 31;
  const int wave = threadIdx.x >> 5;
  const int hi = lane >> 4, lr = lane & 15;
  const int bh = blockIdx.z;
  const int b = bh >> 4, h = bh & 15;
  const __bf16* Qb = Q + (size_t)b * SEQn * Hn + h * HDn;
  const __bf16* Kb = Km + (size_t)b * SEQn * Hn + h * HDn;
  float* Sb = S + (size_t)bh * SEQn * SEQn;
  const int qrow = blockIdx.y * 128 + wave * 16 + lr;
  const int colBase = blockIdx.x * 64;

  v8f acc[4] = {};
#pragma unroll
  for (int k0 = 0; k0 < HDn; k0 += 32) {
    v16bf af = load_frag(Qb + (size_t)qrow * Hn + k0 + hi * 8);
#pragma unroll
    for (int t = 0; t < 4; ++t) {
      v16bf bf = load_frag(Kb + (size_t)(colBase + t * 16 + lr) * Hn + k0 + hi * 8);
      acc[t] = __builtin_amdgcn_wmma_f32_16x16x32_bf16(false, af, false, bf,
                                                       (short)0, acc[t], false, false);
    }
  }
  const int crow = blockIdx.y * 128 + wave * 16 + hi * 8;
#pragma unroll
  for (int t = 0; t < 4; ++t) {
    const int col = colBase + t * 16 + lr;
    const bool pad = kpad && kpad[b * SEQn + col];
#pragma unroll
    for (int r = 0; r < 8; ++r) {
      const int qq = crow + r;
      float v = acc[t][r] * SCALEF;
      if (pad || (causal && col > qq)) v = -INFINITY;
      Sb[(size_t)qq * SEQn + col] = v;
    }
  }
}

// ---------------- O[b,q,h,:] = P[bh,q,:] @ V[b,:,h,:]  (contraction over k) -
__global__ void pav_wmma(const __bf16* __restrict__ P, const __bf16* __restrict__ V,
                         __bf16* __restrict__ O) {
  const int lane = threadIdx.x & 31;
  const int wave = threadIdx.x >> 5;
  const int hi = lane >> 4, lr = lane & 15;
  const int bh = blockIdx.z;
  const int b = bh >> 4, h = bh & 15;
  const __bf16* Pb = P + (size_t)bh * SEQn * SEQn;
  const __bf16* Vb = V + (size_t)b * SEQn * Hn + h * HDn;
  __bf16* Ob = O + (size_t)b * SEQn * Hn + h * HDn;
  const int prow = blockIdx.y * 128 + wave * 16 + lr;

  v8f acc[4] = {};
  for (int k0 = 0; k0 < SEQn; k0 += 32) {
    v16bf af = load_frag(Pb + (size_t)prow * SEQn + k0 + hi * 8);
#pragma unroll
    for (int t = 0; t < 4; ++t) {
      // B operand: lane = output dim d = t*16+lr, K strides down rows of V (ld=Hn)
      v16bf bf = load_frag_strided(Vb + (size_t)(k0 + hi * 8) * Hn + t * 16 + lr, Hn);
      acc[t] = __builtin_amdgcn_wmma_f32_16x16x32_bf16(false, af, false, bf,
                                                       (short)0, acc[t], false, false);
    }
  }
  const int crow = blockIdx.y * 128 + wave * 16 + hi * 8;
#pragma unroll
  for (int t = 0; t < 4; ++t) {
    const int col = t * 16 + lr;
#pragma unroll
    for (int r = 0; r < 8; ++r)
      Ob[(size_t)(crow + r) * Hn + col] = (__bf16)acc[t][r];
  }
}

// ---------------- embedding + sinusoidal positional encoding ---------------
__global__ void embed_pe(const int* __restrict__ tok, const float* __restrict__ emb,
                         float* __restrict__ out) {
  const int bs = blockIdx.x;            // b*SEQ + s
  const int s = bs & (SEQn - 1);
  const int t = tok[bs];
#pragma unroll
  for (int i = 0; i < 4; ++i) {
    const int c = threadIdx.x + i * 256;
    const int half = c >> 1;
    const float div = __expf((float)(2 * half) * (-9.210340371976184f / (float)Hn)); // -ln(1e4)/H
    const float ang = (float)s * div;
    const float pe = (c & 1) ? __cosf(ang) : __sinf(ang);
    out[(size_t)bs * Hn + c] = emb[(size_t)t * Hn + c] + pe;
  }
}

// ---------------- LayerNorm over H=1024, one block(256) per row -------------
// Output is bf16: consumed only as GEMM A-operands.
__global__ void layernorm(const float* __restrict__ x, const float* __restrict__ w,
                          const float* __restrict__ b, __bf16* __restrict__ out) {
  __shared__ float sm[256];
  const int row = blockIdx.x;
  const float* p = x + (size_t)row * Hn;
  float v4[4];
  float s = 0.0f;
#pragma unroll
  for (int i = 0; i < 4; ++i) { v4[i] = p[threadIdx.x + i * 256]; s += v4[i]; }
  sm[threadIdx.x] = s; __syncthreads();
  for (int o = 128; o > 0; o >>= 1) {
    if (threadIdx.x < o) sm[threadIdx.x] += sm[threadIdx.x + o];
    __syncthreads();
  }
  const float mean = sm[0] * (1.0f / Hn);
  __syncthreads();
  float vs = 0.0f;
#pragma unroll
  for (int i = 0; i < 4; ++i) { const float d = v4[i] - mean; vs += d * d; }
  sm[threadIdx.x] = vs; __syncthreads();
  for (int o = 128; o > 0; o >>= 1) {
    if (threadIdx.x < o) sm[threadIdx.x] += sm[threadIdx.x + o];
    __syncthreads();
  }
  const float rstd = rsqrtf(sm[0] * (1.0f / Hn) + 1e-5f);
#pragma unroll
  for (int i = 0; i < 4; ++i) {
    const int c = threadIdx.x + i * 256;
    out[(size_t)row * Hn + c] = (__bf16)((v4[i] - mean) * rstd * w[c] + b[c]);
  }
}

// ---------------- RoPE on [B,S,NH,HD]; fp32 in -> bf16 out ------------------
__global__ void rope_bf(const float* __restrict__ x, __bf16* __restrict__ o) {
  const int idx = blockIdx.x * blockDim.x + threadIdx.x;   // B*S*NH*32 threads
  const int d = idx & 31;
  const int rest = idx >> 5;                               // b*S*NH + s*NH + h
  const int s = (rest >> 4) & (SEQn - 1);                  // NH=16
  const float* p = x + (size_t)rest * HDn;
  __bf16* po = o + (size_t)rest * HDn;
  const float x0 = p[d], x1 = p[d + 32];
  const int i0 = d >> 1;
  const float c0 = -0.28782308510830164f;                  // -2*ln(1e4)/64
  const float th0 = __expf((float)i0 * c0);
  const float th1 = __expf((float)(i0 + 16) * c0);
  const float a0 = (float)s * th0, a1 = (float)s * th1;
  po[d]      = (__bf16)(x0 * __cosf(a0) - x1 * __sinf(a0));
  po[d + 32] = (__bf16)(x1 * __cosf(a1) + x0 * __sinf(a1));
}

// ---------------- softmax: one wave per row (fp32 in -> bf16 probs) ---------
__global__ void softmax_rows(const float* __restrict__ x, __bf16* __restrict__ pr) {
  const int row = blockIdx.x * (blockDim.x >> 5) + (threadIdx.x >> 5);
  const int lane = threadIdx.x & 31;
  const float* p = x + (size_t)row * SEQn;
  __bf16* o = pr + (size_t)row * SEQn;
  float mx = -INFINITY;
  float e[SEQn / 32];
#pragma unroll
  for (int i = 0; i < SEQn / 32; ++i) { e[i] = p[lane + i * 32]; mx = fmaxf(mx, e[i]); }
#pragma unroll
  for (int o2 = 16; o2 > 0; o2 >>= 1) mx = fmaxf(mx, __shfl_xor(mx, o2, 32));
  float sum = 0.0f;
#pragma unroll
  for (int i = 0; i < SEQn / 32; ++i) { e[i] = __expf(e[i] - mx); sum += e[i]; }
#pragma unroll
  for (int o2 = 16; o2 > 0; o2 >>= 1) sum += __shfl_xor(sum, o2, 32);
  const float inv = 1.0f / sum;
#pragma unroll
  for (int i = 0; i < SEQn / 32; ++i) o[lane + i * 32] = (__bf16)(e[i] * inv);
}

// ---------------- GeGLU: out[m,j] = h[m,j] * gelu(h[m,4096+j]) --------------
__global__ void geglu(const float* __restrict__ h, __bf16* __restrict__ out) {
  const int idx = blockIdx.x * blockDim.x + threadIdx.x;   // M*4096 threads
  const int m = idx >> 12, j = idx & (FFHALF - 1);
  const float a = h[(size_t)m * FFNn + j];
  const float g = h[(size_t)m * FFNn + FFHALF + j];
  const float ge = 0.5f * g * (1.0f + erff(g * 0.7071067811865476f));
  out[idx] = (__bf16)(a * ge);
}

// ===========================================================================
extern "C" void kernel_launch(void* const* d_in, const int* in_sizes, int n_in,
                              void* d_out, int out_size, void* d_ws, size_t ws_size,
                              hipStream_t stream) {
  (void)in_sizes; (void)n_in; (void)out_size; (void)ws_size;
  struct LinP  { const __bf16* w; const float* b; };   // w = bf16 copy in ws
  struct AttnP { LinP q, k, v, o; };
  struct LNP   { const float *w, *b; };
  struct FFP   { LinP l1, l2; };
  struct EncL  { AttnP attn; LNP n1; FFP ff; LNP n2; };
  struct DecL  { AttnP sa; LNP n1; AttnP ca; LNP n2; FFP ff; LNP n3; };

  // ---- workspace layout ----
  const size_t A2 = (size_t)Mn * Hn;                    // 2M elems
  float* wsf = (float*)d_ws;
  float* x      = wsf;                                  // fp32 residual stream
  float* qf     = x    + A2;                            // fp32 q before rope
  float* kf     = qf   + A2;                            // fp32 k before rope
  float* ffh    = kf   + A2;                            // [M, 8192] fp32
  float* sc     = ffh  + (size_t)Mn * FFNn;             // [B*NH,512,512] fp32
  __bf16* wsb   = (__bf16*)(sc + (size_t)Bn * NHn * SEQn * SEQn);
  __bf16* h_bf  = wsb;                                  // LN output
  __bf16* q_bf  = h_bf  + A2;
  __bf16* k_bf  = q_bf  + A2;
  __bf16* v_bf  = k_bf  + A2;
  __bf16* ao_bf = v_bf  + A2;
  __bf16* mem_bf= ao_bf + A2;
  __bf16* ffg_bf= mem_bf+ A2;                           // [M, 4096]
  __bf16* pb    = ffg_bf+ (size_t)Mn * FFHALF;          // probs [B*NH,512,512]
  __bf16* wheap = pb    + (size_t)Bn * NHn * SEQn * SEQn; // bf16 weight heap

  // ---- parse inputs; convert every weight matrix to bf16 once ----
  int idx = 0;
  const int* source           = (const int*)d_in[idx++];
  const unsigned char* s_pad  = (const unsigned char*)d_in[idx++];
  const int* target           = (const int*)d_in[idx++];
  const unsigned char* t_pad  = (const unsigned char*)d_in[idx++];
  const float* in_emb         = (const float*)d_in[idx++];
  const float* out_emb        = (const float*)d_in[idx++];

  auto cvtW = [&](const float* src, size_t n) -> const __bf16* {
    __bf16* dst = wheap; wheap += n;
    cvt_f32_bf16<<<(unsigned)(n / 1024), 256, 0, stream>>>(src, dst);   // n % 1024 == 0
    return dst;
  };
  auto getLin = [&](size_t rows, size_t cols) {
    LinP l;
    l.w = cvtW((const float*)d_in[idx], rows * cols); idx++;
    l.b = (const float*)d_in[idx++];
    return l;
  };
  auto getAttn = [&]() { AttnP a; a.q = getLin(Hn, Hn); a.k = getLin(Hn, Hn);
                         a.v = getLin(Hn, Hn); a.o = getLin(Hn, Hn); return a; };
  auto getLN = [&]() { LNP n; n.w = (const float*)d_in[idx++]; n.b = (const float*)d_in[idx++]; return n; };
  auto getFF = [&]() { FFP f; f.l1 = getLin(FFNn, Hn); f.l2 = getLin(Hn, FFHALF); return f; };

  EncL enc[6];
  for (int l = 0; l < 6; ++l) { enc[l].attn = getAttn(); enc[l].n1 = getLN(); enc[l].ff = getFF(); enc[l].n2 = getLN(); }
  LNP enc_norm = getLN();
  DecL dec[6];
  for (int l = 0; l < 6; ++l) { dec[l].sa = getAttn(); dec[l].n1 = getLN(); dec[l].ca = getAttn();
                                dec[l].n2 = getLN(); dec[l].ff = getFF(); dec[l].n3 = getLN(); }
  LNP dec_norm = getLN();
  LinP proj = getLin(VOUTn, Hn);

  // ---- building blocks ----
  auto gemmF = [&](const __bf16* A, LinP lin, const float* res, float* C, int M, int N, int K) {
    gemm_wmma<float><<<dim3(N / 64, M / 128), 256, 0, stream>>>(A, lin.w, lin.b, res, C, M, N, K);
  };
  auto gemmB = [&](const __bf16* A, LinP lin, __bf16* C, int M, int N, int K) {
    gemm_wmma<__bf16><<<dim3(N / 64, M / 128), 256, 0, stream>>>(A, lin.w, lin.b, nullptr, C, M, N, K);
  };
  auto ln = [&](const float* in, LNP n, __bf16* out) {
    layernorm<<<Mn, 256, 0, stream>>>(in, n.w, n.b, out);
  };
  auto mha = [&](const __bf16* qin, const __bf16* kvin, const AttnP& a,
                 const unsigned char* kpad, int causal) {
    gemmF(qin,  a.q, nullptr, qf, Mn, Hn, Hn);
    gemmF(kvin, a.k, nullptr, kf, Mn, Hn, Hn);
    gemmB(kvin, a.v, v_bf, Mn, Hn, Hn);
    rope_bf<<<(Mn * NHn * 32) / 256, 256, 0, stream>>>(qf, q_bf);
    rope_bf<<<(Mn * NHn * 32) / 256, 256, 0, stream>>>(kf, k_bf);
    scores_wmma<<<dim3(SEQn / 64, SEQn / 128, Bn * NHn), 256, 0, stream>>>(q_bf, k_bf, sc, kpad, causal);
    softmax_rows<<<(Bn * NHn * SEQn) / 8, 256, 0, stream>>>(sc, pb);
    pav_wmma<<<dim3(1, SEQn / 128, Bn * NHn), 256, 0, stream>>>(pb, v_bf, ao_bf);
    gemmF(ao_bf, a.o, x, x, Mn, Hn, Hn);                // fused residual into x
  };
  auto ffn = [&](const __bf16* hin, const FFP& f) {
    gemmF(hin, f.l1, nullptr, ffh, Mn, FFNn, Hn);
    geglu<<<((size_t)Mn * FFHALF) / 256, 256, 0, stream>>>(ffh, ffg_bf);
    gemmF(ffg_bf, f.l2, x, x, Mn, Hn, FFHALF);          // fused residual into x
  };

  // ----- encoder -----
  embed_pe<<<Mn, 256, 0, stream>>>(source, in_emb, x);
  for (int l = 0; l < 6; ++l) {
    ln(x, enc[l].n1, h_bf);
    mha(h_bf, h_bf, enc[l].attn, s_pad, 0);
    ln(x, enc[l].n2, h_bf);
    ffn(h_bf, enc[l].ff);
  }
  ln(x, enc_norm, mem_bf);

  // ----- decoder -----
  embed_pe<<<Mn, 256, 0, stream>>>(target, out_emb, x);
  for (int l = 0; l < 6; ++l) {
    ln(x, dec[l].n1, h_bf);
    mha(h_bf, h_bf, dec[l].sa, t_pad, 1);               // causal self-attention
    ln(x, dec[l].n2, h_bf);
    mha(h_bf, mem_bf, dec[l].ca, s_pad, 0);             // cross-attention
    ln(x, dec[l].n3, h_bf);
    ffn(h_bf, dec[l].ff);
  }
  ln(x, dec_norm, h_bf);

  // ----- vocab projection into d_out -----
  gemm_wmma<float><<<dim3(VOUTn / 64, Mn / 128), 256, 0, stream>>>(
      h_bf, proj.w, proj.b, nullptr, (float*)d_out, Mn, VOUTn, Hn);
}
